// LatentDenoisingLayer_56023553409113
// MI455X (gfx1250) — matrix operand order repaired and checked by reading, not compiled, not run on hardware
//
#include <hip/hip_runtime.h>
#include <hip/hip_bf16.h>

#define NN    10000
#define EE    160000
#define CC    32
#define DEF   64
#define HH    8
#define MNODE 9
#define MRES  25

typedef __attribute__((ext_vector_type(16))) _Float16 v16h;
typedef __attribute__((ext_vector_type(8)))  float    v8f;

// ---------------------------------------------------------------- WMMA core
__device__ __forceinline__ v8f wmma_f16(v16h a, v16h b, v8f c) {
  return __builtin_amdgcn_wmma_f32_16x16x32_f16(false, a, false, b, (short)0, c,
                                                false, false);
}

// A fragment (16x32 f16) from a per-lane f32 row pointer. row = lane&15.
// ISA layout: lanes 0-15 hold K=0..7 then 16..23; lanes 16-31 K=8..15 then 24..31.
__device__ __forceinline__ v16h a_frag_global(const float* rowPtr, int k0, int half) {
  v16h a;
  const float* p = rowPtr + k0 + half * 8;
#pragma unroll
  for (int i = 0; i < 8; ++i) {
    a[i]     = (_Float16)p[i];
    a[i + 8] = (_Float16)p[16 + i];
  }
  return a;
}

__device__ __forceinline__ v16h a_frag_lds(const _Float16* As, int ld, int row,
                                           int k0, int half) {
  v16h a;
  const _Float16* p = As + row * ld + k0 + half * 8;
#pragma unroll
  for (int i = 0; i < 8; ++i) {
    a[i]     = p[i];
    a[i + 8] = p[16 + i];
  }
  return a;
}

// B fragment (32x16 f16) from row-major f32 weights W (K x ld).
// lanes 0-15: K = k0+0..7 then k0+16..23, lanes 16-31: K = k0+8..15 then k0+24..31, N = lane&15.
__device__ __forceinline__ v16h b_frag(const float* W, int ld, int k0, int col,
                                       int half) {
  v16h b;
  const float* p = W + (size_t)(k0 + half * 8) * ld + col;
#pragma unroll
  for (int i = 0; i < 8; ++i) {
    b[i]     = (_Float16)p[(size_t)i * ld];
    b[i + 8] = (_Float16)p[(size_t)(16 + i) * ld];
  }
  return b;
}

// fast nonlinearities: v_rcp_f32 instead of IEEE division sequence
__device__ __forceinline__ float sigmoidf_(float x) {
  return __builtin_amdgcn_rcpf(1.f + __expf(-x));
}
__device__ __forceinline__ float siluf_(float x) { return x * sigmoidf_(x); }

// ordered-uint encoding for float atomicMax
__device__ __forceinline__ unsigned fenc(float f) {
  unsigned u = __float_as_uint(f);
  return (u & 0x80000000u) ? ~u : (u | 0x80000000u);
}
__device__ __forceinline__ float fdec(unsigned u) {
  u = (u & 0x80000000u) ? (u & 0x7fffffffu) : ~u;
  return __uint_as_float(u);
}

// ---------------------------------------------------------------- degree
__global__ void deg_kernel(const int* __restrict__ dst, float* __restrict__ deg) {
  int e = blockIdx.x * 256 + threadIdx.x;
  if (e < EE) atomicAdd(&deg[dst[e]], 1.f);
}
__global__ void dnorm_kernel(float* __restrict__ deg) {
  int n = blockIdx.x * 256 + threadIdx.x;
  if (n < NN) deg[n] = rsqrtf(deg[n] + 1.f);
}

// ------------------------------------------------- gate MLP: sigmoid(silu(ef@Wa+ba)@Wb+bb)
// 4 waves/block, one 16-edge tile per wave. E/16 = 10000 tiles -> 2500 blocks.
__global__ void gate_mlp_kernel(const float* __restrict__ ef,
                                const float* __restrict__ Wa, const float* __restrict__ ba,
                                const float* __restrict__ Wb, const float* __restrict__ bb,
                                float* __restrict__ out) {
  __shared__ _Float16 hid[4][16 * 64];
  int w = threadIdx.x >> 5, lane = threadIdx.x & 31;
  int half = lane >> 4, l = lane & 15;
  int tile = blockIdx.x * 4 + w;
  int e0 = tile * 16;
  const float* arow = ef + (size_t)(e0 + l) * DEF;
  for (int ct = 0; ct < 4; ++ct) {
    v8f acc;
    float bv = ba[ct * 16 + l];
#pragma unroll
    for (int i = 0; i < 8; ++i) acc[i] = bv;
    for (int k0 = 0; k0 < DEF; k0 += 32)
      acc = wmma_f16(a_frag_global(arow, k0, half), b_frag(Wa, DEF, k0, ct * 16 + l, half), acc);
#pragma unroll
    for (int i = 0; i < 8; ++i)
      hid[w][(half * 8 + i) * 64 + ct * 16 + l] = (_Float16)siluf_(acc[i]);
  }
  __syncthreads();
  for (int ct = 0; ct < CC / 16; ++ct) {
    v8f acc;
    float bv = bb[ct * 16 + l];
#pragma unroll
    for (int i = 0; i < 8; ++i) acc[i] = bv;
    for (int k0 = 0; k0 < 64; k0 += 32)
      acc = wmma_f16(a_frag_lds(hid[w], 64, l, k0, half), b_frag(Wb, CC, k0, ct * 16 + l, half), acc);
#pragma unroll
    for (int i = 0; i < 8; ++i)
      out[(size_t)(e0 + half * 8 + i) * CC + ct * 16 + l] = sigmoidf_(acc[i]);
  }
}

// -------------------------------- msg = (feat[src][:, :9] @ W) * gate ; atomicAdd -> accum[dst]
// rows = E*9 (90000 tiles) x 2 col tiles -> 180000 waves, 45000 blocks of 4 waves.
__global__ void msg_scatter_kernel(const float* __restrict__ feat, int featStride,
                                   const int* __restrict__ src, const int* __restrict__ dst,
                                   const float* __restrict__ W,
                                   const float* __restrict__ gate,
                                   float* __restrict__ accum) {
  int w = threadIdx.x >> 5, lane = threadIdx.x & 31;
  int half = lane >> 4, l = lane & 15;
  long wid = (long)blockIdx.x * 4 + w;
  int ct = (int)(wid & 1);
  int rt = (int)(wid >> 1);
  int g = rt * 16 + l;
  int e = g / 9, m = g - e * 9;
  const float* arow = feat + (size_t)src[e] * featStride + m * CC;
  v8f acc = {};
  acc = wmma_f16(a_frag_global(arow, 0, half), b_frag(W, CC, 0, ct * 16 + l, half), acc);
  int col = ct * 16 + l;
#pragma unroll
  for (int i = 0; i < 8; ++i) {
    int gi = rt * 16 + half * 8 + i;
    int ei = gi / 9, mi = gi - ei * 9;
    float v = acc[i] * gate[(size_t)ei * CC + col];
    atomicAdd(&accum[((size_t)dst[ei] * 9 + mi) * CC + col], v);
  }
}

// ------------------------------------- x2 = concat(proj * dnorm, node)  (N,9,64)
__global__ void x2_kernel(const float* __restrict__ proj, const float* __restrict__ node,
                          const float* __restrict__ dnorm, float* __restrict__ x2) {
  long idx = (long)blockIdx.x * 256 + threadIdx.x;
  if (idx >= (long)NN * 9 * 64) return;
  int c = (int)(idx & 63);
  long rm = idx >> 6;
  int n = (int)(rm / 9);
  x2[idx] = (c < 32) ? proj[rm * 32 + c] * dnorm[n] : node[rm * 32 + (c - 32)];
}

// ------------------------------------- q = inv @ Wq   (625 row tiles x 8 col tiles)
__global__ void q_kernel(const float* __restrict__ x2, const float* __restrict__ Wq,
                         float* __restrict__ q) {
  int w = threadIdx.x >> 5, lane = threadIdx.x & 31;
  int half = lane >> 4, l = lane & 15;
  long wid = (long)blockIdx.x * 4 + w;
  int ct = (int)(wid & 7);
  int rt = (int)(wid >> 3);
  const float* arow = x2 + (size_t)(rt * 16 + l) * (9 * 64);
  v8f acc = {};
  for (int k0 = 0; k0 < 64; k0 += 32)
    acc = wmma_f16(a_frag_global(arow, k0, half), b_frag(Wq, 128, k0, ct * 16 + l, half), acc);
#pragma unroll
  for (int i = 0; i < 8; ++i)
    q[(size_t)(rt * 16 + half * 8 + i) * 128 + ct * 16 + l] = acc[i];
}

// ----------------- k = concat(inv[src], ef) @ Wk ; logit = <q[dst], k>/4 ; segment max
// 10000 tiles x 8 heads = 80000 waves, 20000 blocks of 4 waves.
__global__ void k_logit_kernel(const float* __restrict__ x2, const float* __restrict__ ef,
                               const float* __restrict__ Wk,
                               const int* __restrict__ src, const int* __restrict__ dst,
                               const float* __restrict__ q,
                               float* __restrict__ logit, unsigned* __restrict__ mEnc) {
  __shared__ float kt[4][16 * 16];
  int w = threadIdx.x >> 5, lane = threadIdx.x & 31;
  int half = lane >> 4, l = lane & 15;
  long wid = (long)blockIdx.x * 4 + w;
  int h = (int)(wid & 7);
  int rt = (int)(wid >> 3);
  int e0 = rt * 16;
  int eL = e0 + l;
  const float* invRow = x2 + (size_t)src[eL] * (9 * 64);
  const float* efRow  = ef + (size_t)eL * DEF;
  v8f acc = {};
  for (int k0 = 0; k0 < 128; k0 += 32) {
    const float* base = (k0 < 64) ? (invRow + k0) : (efRow + (k0 - 64));
    acc = wmma_f16(a_frag_global(base, 0, half), b_frag(Wk, 128, k0, h * 16 + l, half), acc);
  }
#pragma unroll
  for (int i = 0; i < 8; ++i) kt[w][(half * 8 + i) * 16 + l] = acc[i];
  __syncthreads();
  if (lane < 16) {
    int e = e0 + lane;
    int d = dst[e];
    const float* qp = q + (size_t)d * 128 + h * 16;
    float s = 0.f;
#pragma unroll
    for (int a2 = 0; a2 < 16; ++a2) s += qp[a2] * kt[w][lane * 16 + a2];
    s *= 0.25f;  // 1/sqrt(16)
    logit[(size_t)e * HH + h] = s;
    atomicMax(&mEnc[(size_t)d * HH + h], fenc(s));
  }
}

// ------------------------------------- z = exp(logit - m[dst]); den += z (in-place z)
__global__ void zden_kernel(float* __restrict__ logit, const unsigned* __restrict__ mEnc,
                            const int* __restrict__ dst, float* __restrict__ den) {
  long idx = (long)blockIdx.x * 256 + threadIdx.x;
  if (idx >= (long)EE * HH) return;
  int e = (int)(idx >> 3), h = (int)(idx & 7);
  int d = dst[e];
  float z = __expf(logit[idx] - fdec(mEnc[(size_t)d * HH + h]));
  logit[idx] = z;
  atomicAdd(&den[(size_t)d * HH + h], z);
}

// ------------ v = x2[src] @ Wv ; scaled by alpha ; atomicAdd -> vagg[dst]  (fused, never stored)
// rows = E*9 (90000 tiles) x 8 col tiles -> 720000 waves, 180000 blocks.
__global__ void v_scatter_kernel(const float* __restrict__ x2, const float* __restrict__ Wv,
                                 const int* __restrict__ src, const int* __restrict__ dst,
                                 const float* __restrict__ z, const float* __restrict__ den,
                                 float* __restrict__ vagg) {
  int w = threadIdx.x >> 5, lane = threadIdx.x & 31;
  int half = lane >> 4, l = lane & 15;
  long wid = (long)blockIdx.x * 4 + w;
  int ct = (int)(wid & 7);
  long rt = wid >> 3;
  int g = (int)(rt * 16 + l);
  int e = g / 9, m = g - e * 9;
  const float* arow = x2 + (size_t)src[e] * (9 * 64) + m * 64;
  v8f acc = {};
  for (int k0 = 0; k0 < 64; k0 += 32)
    acc = wmma_f16(a_frag_global(arow, k0, half), b_frag(Wv, 128, k0, ct * 16 + l, half), acc);
  int col = ct * 16 + l;
#pragma unroll
  for (int i = 0; i < 8; ++i) {
    int gi = (int)(rt * 16 + half * 8 + i);
    int ei = gi / 9, mi = gi - ei * 9;
    int d = dst[ei];
    float alpha = z[(size_t)ei * HH + ct] *
                  __builtin_amdgcn_rcpf(den[(size_t)d * HH + ct] + 1e-9f);
    atomicAdd(&vagg[((size_t)d * 9 + mi) * 128 + col], acc[i] * alpha);
  }
}

// ------------------------------------- hbuf = node + vagg @ Wo   (5625 x 2 tiles, 2 waves/blk)
__global__ void attn_out_kernel(const float* __restrict__ vagg, const float* __restrict__ Wo,
                                const float* __restrict__ node, float* __restrict__ hbuf) {
  int w = threadIdx.x >> 5, lane = threadIdx.x & 31;
  int half = lane >> 4, l = lane & 15;
  long wid = (long)blockIdx.x * 2 + w;
  int ct = (int)(wid & 1);
  long rt = wid >> 1;
  const float* arow = vagg + (size_t)(rt * 16 + l) * 128;
  v8f acc = {};
  for (int k0 = 0; k0 < 128; k0 += 32)
    acc = wmma_f16(a_frag_global(arow, k0, half), b_frag(Wo, 32, k0, ct * 16 + l, half), acc);
#pragma unroll
  for (int i = 0; i < 8; ++i) {
    long r = rt * 16 + half * 8 + i;
    long o = r * 32 + ct * 16 + l;
    hbuf[o] = node[o] + acc[i];
  }
}

// ------------------------------------- SO(3) RMS norm (one wave per node, lane = channel)
__global__ void norm_kernel(float* __restrict__ hbuf, const float* __restrict__ gamma) {
  int w = threadIdx.x >> 5, lane = threadIdx.x & 31;
  int n = blockIdx.x * 4 + w;
  if (n >= NN) return;
  float* base = hbuf + (size_t)n * 9 * 32;
  const int s0[3] = {0, 1, 4}, s1[3] = {1, 4, 9};
  for (int sl = 0; sl < 3; ++sl) {
    int rs = s0[sl], re = s1[sl];
    float vv[5];
    float sum = 0.f;
    for (int m = rs; m < re; ++m) {
      float v = base[m * 32 + lane];
      vv[m - rs] = v;
      sum += v * v;
    }
    for (int off = 16; off; off >>= 1) sum += __shfl_xor(sum, off, 32);
    // gamma / rms  ->  gamma * rsqrt(mean + eps), single v_rsq_f32
    float inv = rsqrtf(sum / ((re - rs) * 32) + 1e-6f);
    float gm = gamma[sl * 32 + lane] * inv;
    for (int m = rs; m < re; ++m) base[m * 32 + lane] = vv[m - rs] * gm;
  }
}

// ------------------------------------- g = silu(h[:,0,:] @ Wg + bg)  (625 x 4 tiles)
__global__ void ggate_kernel(const float* __restrict__ hbuf, const float* __restrict__ Wg,
                             const float* __restrict__ bg, float* __restrict__ gbuf) {
  int w = threadIdx.x >> 5, lane = threadIdx.x & 31;
  int half = lane >> 4, l = lane & 15;
  long wid = (long)blockIdx.x * 4 + w;
  int ct = (int)(wid & 3);
  int rt = (int)(wid >> 2);
  const float* arow = hbuf + (size_t)(rt * 16 + l) * (9 * 32);
  v8f acc;
  float bv = bg[ct * 16 + l];
#pragma unroll
  for (int i = 0; i < 8; ++i) acc[i] = bv;
  acc = wmma_f16(a_frag_global(arow, 0, half), b_frag(Wg, 64, 0, ct * 16 + l, half), acc);
#pragma unroll
  for (int i = 0; i < 8; ++i)
    gbuf[(size_t)(rt * 16 + half * 8 + i) * 64 + ct * 16 + l] = siluf_(acc[i]);
}

// ------------- new_node = (h@W1 * g) @ W2 + h   (one wave per 16-row tile, 5625 blocks)
__global__ void ffn_kernel(const float* __restrict__ hbuf, const float* __restrict__ W1,
                           const float* __restrict__ gbuf, const float* __restrict__ W2,
                           float* __restrict__ out_node) {
  __shared__ _Float16 hid[16 * 64];
  int lane = threadIdx.x & 31, half = lane >> 4, l = lane & 15;
  int rt = blockIdx.x;
  const float* arow = hbuf + (size_t)(rt * 16 + l) * 32;
  for (int ct = 0; ct < 4; ++ct) {
    v8f acc = {};
    acc = wmma_f16(a_frag_global(arow, 0, half), b_frag(W1, 64, 0, ct * 16 + l, half), acc);
#pragma unroll
    for (int i = 0; i < 8; ++i) {
      int r = rt * 16 + half * 8 + i;
      int n = r / 9;
      hid[(half * 8 + i) * 64 + ct * 16 + l] =
          (_Float16)(acc[i] * gbuf[(size_t)n * 64 + ct * 16 + l]);
    }
  }
  __syncthreads();
  for (int ct = 0; ct < 2; ++ct) {
    v8f acc = {};
    for (int k0 = 0; k0 < 64; k0 += 32)
      acc = wmma_f16(a_frag_lds(hid, 64, l, k0, half), b_frag(W2, 32, k0, ct * 16 + l, half), acc);
#pragma unroll
    for (int i = 0; i < 8; ++i) {
      long r = (long)rt * 16 + half * 8 + i;
      long o = r * 32 + ct * 16 + l;
      out_node[o] = acc[i] + hbuf[o];
    }
  }
}

// ------------------------------------- new_res = res + pad(agg2 * dnorm)
__global__ void res_out_kernel(const float* __restrict__ res, const float* __restrict__ proj,
                               const float* __restrict__ dnorm, float* __restrict__ out_res) {
  long idx = (long)blockIdx.x * 256 + threadIdx.x;
  if (idx >= (long)NN * MRES * CC) return;
  int c = (int)(idx & 31);
  long t = idx >> 5;
  int n = (int)(t / MRES), m = (int)(t - (long)n * MRES);
  float v = res[idx];
  if (m < MNODE) v += proj[((long)n * 9 + m) * 32 + c] * dnorm[n];
  out_res[idx] = v;
}

// ------- new_ef = ef + silu(concat(ef, inv[src], inv[dst]) @ We1 + be1) @ We2 + be2
// 10000 tiles, 4 waves/block -> 2500 blocks.
__global__ void edge_mlp_kernel(const float* __restrict__ ef, const float* __restrict__ out_node,
                                const int* __restrict__ src, const int* __restrict__ dst,
                                const float* __restrict__ We1, const float* __restrict__ be1,
                                const float* __restrict__ We2, const float* __restrict__ be2,
                                float* __restrict__ out_ef) {
  __shared__ _Float16 hid[4][16 * 64];
  int w = threadIdx.x >> 5, lane = threadIdx.x & 31;
  int half = lane >> 4, l = lane & 15;
  int tile = blockIdx.x * 4 + w;
  int e0 = tile * 16;
  int eL = e0 + l;
  const float* efRow = ef + (size_t)eL * DEF;
  const float* sRow = out_node + (size_t)src[eL] * (9 * 32);
  const float* dRow = out_node + (size_t)dst[eL] * (9 * 32);
  for (int ct = 0; ct < 4; ++ct) {
    v8f acc;
    float bv = be1[ct * 16 + l];
#pragma unroll
    for (int i = 0; i < 8; ++i) acc[i] = bv;
    for (int k0 = 0; k0 < 128; k0 += 32) {
      const float* base = (k0 < 64) ? (efRow + k0) : ((k0 == 64) ? sRow : dRow);
      acc = wmma_f16(a_frag_global(base, 0, half), b_frag(We1, 64, k0, ct * 16 + l, half), acc);
    }
#pragma unroll
    for (int i = 0; i < 8; ++i)
      hid[w][(half * 8 + i) * 64 + ct * 16 + l] = (_Float16)siluf_(acc[i]);
  }
  __syncthreads();
  for (int ct = 0; ct < 4; ++ct) {
    v8f acc;
    float bv = be2[ct * 16 + l];
#pragma unroll
    for (int i = 0; i < 8; ++i) acc[i] = bv;
    for (int k0 = 0; k0 < 64; k0 += 32)
      acc = wmma_f16(a_frag_lds(hid[w], 64, l, k0, half), b_frag(We2, 64, k0, ct * 16 + l, half), acc);
#pragma unroll
    for (int i = 0; i < 8; ++i) {
      size_t o = (size_t)(e0 + half * 8 + i) * DEF + ct * 16 + l;
      out_ef[o] = ef[o] + acc[i];
    }
  }
}

// ================================================================= launch
extern "C" void kernel_launch(void* const* d_in, const int* in_sizes, int n_in,
                              void* d_out, int out_size, void* d_ws, size_t ws_size,
                              hipStream_t stream) {
  (void)in_sizes; (void)n_in; (void)out_size; (void)ws_size;

  const float* node = (const float*)d_in[0];   // (N,9,32)
  const float* res  = (const float*)d_in[1];   // (N,25,32)
  const float* ef   = (const float*)d_in[2];   // (E,64)
  const int*   ei   = (const int*)d_in[3];     // (2,E)
  const int* src = ei;
  const int* dst = ei + EE;
  const float* g1a = (const float*)d_in[4];
  const float* g1b = (const float*)d_in[5];
  const float* g1c = (const float*)d_in[6];
  const float* g1d = (const float*)d_in[7];
  const float* Wpn = (const float*)d_in[8];
  const float* Wq  = (const float*)d_in[9];
  const float* Wk  = (const float*)d_in[10];
  const float* Wv  = (const float*)d_in[11];
  const float* Wo  = (const float*)d_in[12];
  const float* gamma = (const float*)d_in[13];
  const float* W1  = (const float*)d_in[14];
  const float* Wg  = (const float*)d_in[15];
  const float* bg  = (const float*)d_in[16];
  const float* W2  = (const float*)d_in[17];
  const float* g2a = (const float*)d_in[18];
  const float* g2b = (const float*)d_in[19];
  const float* g2c = (const float*)d_in[20];
  const float* g2d = (const float*)d_in[21];
  const float* Wnp = (const float*)d_in[22];
  const float* We1 = (const float*)d_in[23];
  const float* be1 = (const float*)d_in[24];
  const float* We2 = (const float*)d_in[25];
  const float* be2 = (const float*)d_in[26];

  float* out_node = (float*)d_out;              // N*9*32
  float* out_res  = out_node + (size_t)NN * MNODE * CC;   // N*25*32
  float* out_ef   = out_res + (size_t)NN * MRES * CC;     // E*64

  // workspace layout (floats)
  float* ws = (float*)d_ws;
  float*    dnorm  = ws;                                  // N (pad 16384)
  unsigned* mEnc   = (unsigned*)(ws + 16384);             // N*8 (pad 81920)
  float*    den    = ws + 16384 + 81920;                  // N*8 (pad 81920)
  float*    gateB  = den + 81920;                         // E*32
  float*    proj   = gateB + (size_t)EE * CC;             // N*9*32
  float*    x2     = proj + (size_t)NN * 9 * 32;          // N*9*64
  float*    qbuf   = x2 + (size_t)NN * 9 * 64;            // N*128
  float*    logitB = qbuf + (size_t)NN * 128;             // E*8
  float*    vagg   = logitB + (size_t)EE * HH;            // N*9*128
  float*    hbuf   = vagg + (size_t)NN * 9 * 128;         // N*9*32
  float*    gbuf   = hbuf + (size_t)NN * 9 * 32;          // N*64

  // zero-init accumulators (graph-capture safe)
  hipMemsetAsync(dnorm, 0, (size_t)NN * 4, stream);
  hipMemsetAsync(mEnc, 0, (size_t)NN * HH * 4, stream);
  hipMemsetAsync(den, 0, (size_t)NN * HH * 4, stream);
  hipMemsetAsync(proj, 0, (size_t)NN * 9 * 32 * 4, stream);
  hipMemsetAsync(vagg, 0, (size_t)NN * 9 * 128 * 4, stream);

  // 1. degree / dnorm
  deg_kernel<<<(EE + 255) / 256, 256, 0, stream>>>(dst, dnorm);
  dnorm_kernel<<<(NN + 255) / 256, 256, 0, stream>>>(dnorm);

  // 2. gate1
  gate_mlp_kernel<<<EE / 16 / 4, 128, 0, stream>>>(ef, g1a, g1b, g1c, g1d, gateB);

  // 3. msg1 scatter into proj
  msg_scatter_kernel<<<(EE * 9 / 16) * 2 / 4, 128, 0, stream>>>(
      res, MRES * CC, src, dst, Wpn, gateB, proj);

  // 4. x2 = concat(proj*dnorm, node)
  x2_kernel<<<(NN * 9 * 64) / 256, 256, 0, stream>>>(proj, node, dnorm, x2);

  // 5. q
  q_kernel<<<(NN / 16) * 8 / 4, 128, 0, stream>>>(x2, Wq, qbuf);

  // 6. k + logits + segment max
  k_logit_kernel<<<(EE / 16) * 8 / 4, 128, 0, stream>>>(x2, ef, Wk, src, dst, qbuf,
                                                        logitB, mEnc);

  // 7. z / den
  zden_kernel<<<(EE * HH) / 256, 256, 0, stream>>>(logitB, mEnc, dst, den);

  // 8. v projection fused with alpha scaling + scatter
  v_scatter_kernel<<<(EE * 9 / 16) * 8 / 4, 128, 0, stream>>>(x2, Wv, src, dst,
                                                              logitB, den, vagg);

  // 9. attention output projection + residual
  attn_out_kernel<<<(NN * 9 / 16) * 2 / 2, 64, 0, stream>>>(vagg, Wo, node, hbuf);

  // 10. SO(3) norm (in place on hbuf)
  norm_kernel<<<NN / 4, 128, 0, stream>>>(hbuf, gamma);

  // 11. gated FFN
  ggate_kernel<<<(NN / 16) * 4 / 4, 128, 0, stream>>>(hbuf, Wg, bg, gbuf);
  ffn_kernel<<<NN * 9 / 16, 32, 0, stream>>>(hbuf, W1, gbuf, W2, out_node);

  // 12. gate2 + msg2 scatter (reuse gateB, proj)
  gate_mlp_kernel<<<EE / 16 / 4, 128, 0, stream>>>(ef, g2a, g2b, g2c, g2d, gateB);
  hipMemsetAsync(proj, 0, (size_t)NN * 9 * 32 * 4, stream);
  msg_scatter_kernel<<<(EE * 9 / 16) * 2 / 4, 128, 0, stream>>>(
      out_node, MNODE * CC, src, dst, Wnp, gateB, proj);

  // 13. residual-stream output
  res_out_kernel<<<(NN * MRES * CC) / 256, 256, 0, stream>>>(res, proj, dnorm, out_res);

  // 14. edge-feature MLP
  edge_mlp_kernel<<<EE / 16 / 4, 128, 0, stream>>>(ef, out_node, src, dst, We1, be1,
                                                   We2, be2, out_ef);
}
